// FrameAugment_39771397161405
// MI455X (gfx1250) — matrix operand order repaired and checked by reading, not compiled, not run on hardware
//
#include <hip/hip_runtime.h>

typedef __attribute__((ext_vector_type(16))) _Float16 v16h;
typedef __attribute__((ext_vector_type(8)))  _Float16 v8h;
typedef __attribute__((ext_vector_type(8)))  float    v8f;
typedef __attribute__((ext_vector_type(4)))  float    v4f;

#define B_ 8
#define S_ 2048
#define F_ 128

// ---------------------------------------------------------------- init
__global__ void fa_init_kernel(unsigned int* __restrict__ maxbits) {
    if (threadIdx.x == 0) *maxbits = 0u;
}

// ---------------------------------------------------------------- row stats
// inv-variance per (b,s) row of feature (ddof=1) + global max via atomicMax
// on float bit pattern (values strictly positive -> uint order == float order)
__global__ __launch_bounds__(256)
void fa_stats_kernel(const float* __restrict__ feature,
                     float* __restrict__ stab_raw,
                     unsigned int* __restrict__ maxbits) {
    const int lane = threadIdx.x & 31;
    const int wave = threadIdx.x >> 5;
    const int row  = blockIdx.x * 8 + wave;          // B_*S_ rows total
    const float* fr = feature + (size_t)row * F_;
    float s = 0.f, s2 = 0.f;
#pragma unroll
    for (int i = 0; i < 4; ++i) {
        float v = fr[lane + 32 * i];
        s += v; s2 += v * v;
    }
#pragma unroll
    for (int off = 16; off; off >>= 1) {
        s  += __shfl_xor(s,  off, 32);
        s2 += __shfl_xor(s2, off, 32);
    }
    if (lane == 0) {
        float mean = s * (1.f / F_);
        float var  = (s2 - (float)F_ * mean * mean) * (1.f / (F_ - 1));
        float inv  = 1.f / (var + 1e-6f);
        stab_raw[row] = inv;
        atomicMax(maxbits, __float_as_uint(inv));
    }
}

// ---------------------------------------------------------------- transpose+convert
// featT[b][c][s] = (f16) feature[b][s][c]  (LDS tile: coalesced both ways)
__global__ __launch_bounds__(256)
void fa_cvt_kernel(const float* __restrict__ feature,
                   _Float16* __restrict__ featT) {
    __shared__ _Float16 tile[F_ * 18];               // pad stride 18 vs banks
    const int b  = blockIdx.x / (S_ / 16);
    const int s0 = (blockIdx.x % (S_ / 16)) * 16;
    const int t  = threadIdx.x;
#pragma unroll
    for (int k = 0; k < 8; ++k) {
        int e  = t + k * 256;                        // 0..2047
        int sl = e >> 7;                             // s within tile
        int c  = e & 127;                            // consecutive t -> consecutive c
        tile[c * 18 + sl] =
            (_Float16)feature[((size_t)b * S_ + s0 + sl) * F_ + c];
    }
    __syncthreads();
#pragma unroll
    for (int k = 0; k < 8; ++k) {
        int e  = t + k * 256;
        int c  = e >> 4;
        int sl = e & 15;                             // consecutive t -> consecutive s
        featT[((size_t)b * F_ + c) * S_ + s0 + sl] = tile[c * 18 + sl];
    }
}

// ---------------------------------------------------------------- fused flash pass
// one wave owns 16 output rows x all 2048 keys; online softmax + P@V via WMMA
__global__ __launch_bounds__(128)
void fa_flash_kernel(const float* __restrict__ noise,
                     const _Float16* __restrict__ featT,
                     const float* __restrict__ stab_raw,
                     const unsigned int* __restrict__ maxbits,
                     float* __restrict__ out) {
    const int lane = threadIdx.x & 31;
    const int wave = threadIdx.x >> 5;
    const int rowblock = blockIdx.x * 4 + wave;      // 1024 row blocks
    const int b  = rowblock >> 7;                    // / (S_/16)
    const int r0 = (rowblock & 127) << 4;
    const int rl = lane & 15;                        // my row within block
    const int kh = lane >> 4;                        // K-half (A/B lane layout)

    const float maxv = __uint_as_float(*maxbits);
    const float stab = stab_raw[b * S_ + r0 + rl] / maxv;

    const float* nrow = noise + ((size_t)b * S_ + r0 + rl) * S_;
    const _Float16* fb = featT + (size_t)b * F_ * S_ + (size_t)kh * 8;

    v8f acc[8];
#pragma unroll
    for (int t = 0; t < 8; ++t)
#pragma unroll
        for (int e = 0; e < 8; ++e) acc[t][e] = 0.f;

    float m = -1e30f;
    float l = 0.f;

    for (int k0 = 0; k0 < S_; k0 += 32) {
        if (k0 + 64 < S_) __builtin_prefetch(nrow + k0 + 64, 0, 0);

        // ---- noise tile in 16-bit A-matrix lane layout (16x32) ----
        // low lanes: K = {0..7, 16..23}; high lanes: K = {8..15, 24..31}
        const v4f* p0 = (const v4f*)(nrow + k0 + kh * 8);
        const v4f* p1 = (const v4f*)(nrow + k0 + 16 + kh * 8);
        v4f x0 = p0[0], x1 = p0[1], x2 = p1[0], x3 = p1[1];
        float x[16];
#pragma unroll
        for (int e = 0; e < 4; ++e) {
            x[e]      = x0[e] * stab;
            x[4 + e]  = x1[e] * stab;
            x[8 + e]  = x2[e] * stab;
            x[12 + e] = x3[e] * stab;
        }

        // ---- online softmax: row max across both K-halves ----
        float tm = x[0];
#pragma unroll
        for (int j = 1; j < 16; ++j) tm = fmaxf(tm, x[j]);
        tm = fmaxf(tm, __shfl_xor(tm, 16, 32));
        const float mOld = m;
        const float mn   = fmaxf(mOld, tm);
        m = mn;

        v16h a;
        float ls = 0.f;
#pragma unroll
        for (int j = 0; j < 16; ++j) {
            float pj = __expf(x[j] - mn);
            ls += pj;
            a[j] = (_Float16)pj;
        }

        // ---- rescale only when some row's max actually grew ----
        // steady state (random scores): max stops changing -> scale==1 exactly,
        // skip 32 v_pk_mul_f32 + 8 ds_bpermute. Branch is wave-uniform (__any),
        // so EXEC remains all-ones for the WMMAs below.
        if (__any(mn > mOld)) {
            float scale = __expf(mOld - mn);         // ==1 for unchanged rows
            l = l * scale + ls;
            float sc[8];                             // C layout: vgpr j -> row j|j+8
#pragma unroll
            for (int j = 0; j < 8; ++j) sc[j] = __shfl(scale, j + kh * 8, 32);
#pragma unroll
            for (int t = 0; t < 8; ++t)
#pragma unroll
                for (int j = 0; j < 8; ++j) acc[t][j] *= sc[j];
        } else {
            l += ls;
        }

        // ---- 8 N-tiles of V (f16, 32x16 B layout) + WMMA ----
#pragma unroll
        for (int t = 0; t < 8; ++t) {
            const v8h* bp = (const v8h*)(fb + (size_t)(t * 16 + rl) * S_ + k0);
            v8h lo = bp[0], hi = bp[2];
            v16h bt = __builtin_shufflevector(lo, hi,
                0, 1, 2, 3, 4, 5, 6, 7, 8, 9, 10, 11, 12, 13, 14, 15);
            acc[t] = __builtin_amdgcn_wmma_f32_16x16x32_f16(
                false, a, false, bt, (short)0, acc[t], false, false);
        }
    }

    // ---- finalize: combine row sums across K-halves, normalize, store ----
    l += __shfl_xor(l, 16, 32);
    float rinv[8];
#pragma unroll
    for (int j = 0; j < 8; ++j) rinv[j] = 1.f / __shfl(l, j + kh * 8, 32);

    float* orow = out + ((size_t)b * S_ + r0 + kh * 8) * F_;
#pragma unroll
    for (int t = 0; t < 8; ++t)
#pragma unroll
        for (int j = 0; j < 8; ++j)
            orow[(size_t)j * F_ + t * 16 + rl] = acc[t][j] * rinv[j];
}

// ---------------------------------------------------------------- launch
extern "C" void kernel_launch(void* const* d_in, const int* in_sizes, int n_in,
                              void* d_out, int out_size, void* d_ws, size_t ws_size,
                              hipStream_t stream) {
    const float* feature = (const float*)d_in[0];
    const float* noise   = (const float*)d_in[1];
    float* out = (float*)d_out;

    unsigned int* maxbits = (unsigned int*)d_ws;
    float* stab_raw = (float*)((char*)d_ws + 64);
    _Float16* featT = (_Float16*)((char*)d_ws + 64 + (size_t)B_ * S_ * sizeof(float));

    hipLaunchKernelGGL(fa_init_kernel, dim3(1), dim3(64), 0, stream, maxbits);
    hipLaunchKernelGGL(fa_stats_kernel, dim3(B_ * S_ / 8), dim3(256), 0, stream,
                       feature, stab_raw, maxbits);
    hipLaunchKernelGGL(fa_cvt_kernel, dim3(B_ * (S_ / 16)), dim3(256), 0, stream,
                       feature, featT);
    hipLaunchKernelGGL(fa_flash_kernel, dim3(B_ * (S_ / 16) / 4), dim3(128), 0, stream,
                       noise, featT, stab_raw, maxbits, out);
}
